// MultiheadFlashDiff1_14224931684916
// MI455X (gfx1250) — matrix operand order repaired and checked by reading, not compiled, not run on hardware
//
#include <hip/hip_runtime.h>
#include <hip/hip_bf16.h>

typedef _Float16 f16;
typedef _Float16 v16h __attribute__((ext_vector_type(16)));
typedef _Float16 v8h  __attribute__((ext_vector_type(8)));
typedef float    v8f  __attribute__((ext_vector_type(8)));

#define B_ 2
#define T_ 2048
#define E_ 1024
#define H_ 8
#define D_ 64
#define HD_ 128        // 2*D per head
#define NTOK (B_*T_)   // 4096

#define LAMBDA_INIT 0.35550906759096925f
#define ONE_MINUS_LAMBDA 0.64449093240903075f

static __device__ __forceinline__ v16h cat8(v8h a, v8h b) {
    return __builtin_shufflevector(a, b, 0,1,2,3,4,5,6,7,8,9,10,11,12,13,14,15);
}

// ---------------- fp32 -> f16 convert ----------------
__global__ void cvt_f32_f16(const float* __restrict__ in, f16* __restrict__ out, int n) {
    int i = blockIdx.x * blockDim.x + threadIdx.x;
    if (i < n) out[i] = (f16)in[i];
}

// ---------------- weight transpose + convert: wt[n*E+k] = w[k*E+n] ----------------
__global__ void transpose_cvt_w(const float* __restrict__ w, f16* __restrict__ wt) {
    int i = blockIdx.x * blockDim.x + threadIdx.x;     // over E*E
    int k = i >> 10;            // i / E_
    int n = i & (E_ - 1);       // i % E_
    wt[n * E_ + k] = (f16)w[i];
}

// ---------------- WMMA GEMM: C(MxN) = A(MxK, row-major f16) * BT(NxK, row-major f16)^T -------
// one wave computes a 32x64 output tile (2x4 subtiles of 16x16); 8 waves / block.
// OUTMODE: 0 = f16 row-major, 1 = f32 row-major, 2 = f16 scattered into vt (B,H,128,T) layout
template <int OUTMODE>
__global__ void gemm_wmma(const f16* __restrict__ A, const f16* __restrict__ BT,
                          void* __restrict__ Cout, int M, int N, int K) {
    int lane = threadIdx.x & 31;
    int wave = threadIdx.x >> 5;
    int tile = blockIdx.x * (blockDim.x >> 5) + wave;
    int ntn  = N >> 6;                          // 64-wide n tiles
    int tm = tile / ntn;
    int tn = tile % ntn;
    if (tm >= (M >> 5)) return;                 // wave-uniform
    int m0 = tm << 5, n0 = tn << 6;
    int hi = lane >> 4;
    int lr = lane & 15;

    v8f acc[2][4];
#pragma unroll
    for (int i = 0; i < 2; ++i)
#pragma unroll
        for (int j = 0; j < 4; ++j)
            acc[i][j] = (v8f){0.f,0.f,0.f,0.f,0.f,0.f,0.f,0.f};

    const f16* arow0 = A  + (long)(m0 + lr) * K + (hi ? 16 : 0);
    const f16* arow1 = arow0 + 16 * (long)K;
    const f16* brow  = BT + (long)(n0 + lr) * K + (hi ? 16 : 0);

    for (int k0 = 0; k0 < K; k0 += 32) {
        v16h av[2];
        av[0] = cat8(*(const v8h*)(arow0 + k0), *(const v8h*)(arow0 + k0 + 8));
        av[1] = cat8(*(const v8h*)(arow1 + k0), *(const v8h*)(arow1 + k0 + 8));
        v16h bv[4];
#pragma unroll
        for (int j = 0; j < 4; ++j)
            bv[j] = *(const v16h*)(brow + (long)j * 16 * K + k0);
#pragma unroll
        for (int i = 0; i < 2; ++i)
#pragma unroll
            for (int j = 0; j < 4; ++j)
                acc[i][j] = __builtin_amdgcn_wmma_f32_16x16x32_f16(
                    false, av[i], false, bv[j], (short)0, acc[i][j], false, false);
    }

#pragma unroll
    for (int i = 0; i < 2; ++i) {
#pragma unroll
        for (int j = 0; j < 4; ++j) {
#pragma unroll
            for (int r = 0; r < 8; ++r) {
                int row = m0 + i * 16 + r + hi * 8;
                int col = n0 + j * 16 + lr;
                float val = acc[i][j][r];
                if (OUTMODE == 0) {
                    ((f16*)Cout)[(long)row * N + col] = (f16)val;
                } else if (OUTMODE == 1) {
                    ((float*)Cout)[(long)row * N + col] = val;
                } else {
                    int b = row / T_, t = row % T_;
                    int h = col >> 7, n = col & 127;
                    ((f16*)Cout)[((long)(b * H_ + h) * HD_ + n) * T_ + t] = (f16)val;
                }
            }
        }
    }
}

// ---------------- fused differential flash attention -----------------------------------------
// grid: (T/16, B*H), block: 32 (one wave). q16/k16: (B,T,E) f16 with per-head [h*128 + sel*64 + d]
// vt16: (B,H,128,T) f16.  Output a16: (B,T,E) f16 (normalized differential attention result).
// Row-sum of P is computed by the matrix engine itself: a 9th accumulator tile with an
// all-ones B fragment gives every lane the row sum (no cross-lane shuffle reduction needed).
__global__ void __launch_bounds__(32) diff_attn(
        const f16* __restrict__ q16, const f16* __restrict__ k16,
        const f16* __restrict__ vt16, f16* __restrict__ a16,
        const float* __restrict__ lq1, const float* __restrict__ lk1,
        const float* __restrict__ lq2, const float* __restrict__ lk2,
        const float* __restrict__ g) {
    __shared__ f16   lds_p[16][32];     // P tile staging (C-layout -> A-fragment reshape)
    __shared__ float lds_o1[16][HD_];   // sel=0 normalized output (per-lane private spill store)

    int lane = threadIdx.x & 31;
    int hi = lane >> 4;
    int lr = lane & 15;
    int qt = blockIdx.x;
    int bh = blockIdx.y;
    int b = bh / H_, h = bh % H_;
    int m0 = qt << 4;
    const float scale = 0.125f;   // 1/sqrt(D), D=64

    // lambda scalars (uniform)
    float s1 = 0.f, s2 = 0.f;
    for (int i = 0; i < D_; ++i) { s1 += lq1[i] * lk1[i]; s2 += lq2[i] * lk2[i]; }
    const float lam = __expf(s1) - __expf(s2) + LAMBDA_INIT;

    v16h bones;                         // all-ones B fragment for row-sum tile
#pragma unroll
    for (int j = 0; j < 16; ++j) bones[j] = (f16)1.0f;

#pragma unroll
    for (int sel = 0; sel < 2; ++sel) {
        // Q A-fragments for two 32-wide K chunks over D=64
        const f16* qbase = q16 + (long)(b * T_ + m0 + lr) * E_ + h * HD_ + sel * D_ + (hi ? 16 : 0);
        v16h qa[2];
#pragma unroll
        for (int kk = 0; kk < 2; ++kk)
            qa[kk] = cat8(*(const v8h*)(qbase + kk * 32), *(const v8h*)(qbase + kk * 32 + 8));

        v8f acc[9];                     // [0..7]: output n-tiles, [8]: row-sum tile
#pragma unroll
        for (int t = 0; t < 9; ++t) acc[t] = (v8f){0.f,0.f,0.f,0.f,0.f,0.f,0.f,0.f};
        float mrow[8];
#pragma unroll
        for (int r = 0; r < 8; ++r) mrow[r] = -1e30f;

        // running pointers, stepped by 32 keys per iteration
        const f16* kptr = k16 + (long)(b * T_ + lr) * E_ + h * HD_ + sel * D_ + (hi ? 16 : 0);
        const f16* vptr = vt16 + ((long)(b * H_ + h) * HD_ + lr) * T_ + (hi ? 16 : 0);

        for (int ks = 0; ks < m0 + 16; ks += 32) {      // causal key range, 32 keys/iter
            // batch-load all four K B-fragments (2 column-halves x 2 K-chunks), then 4 WMMAs
            v16h kv00 = *(const v16h*)(kptr);
            v16h kv01 = *(const v16h*)(kptr + 32);
            v16h kv10 = *(const v16h*)(kptr + 16 * (long)E_);
            v16h kv11 = *(const v16h*)(kptr + 16 * (long)E_ + 32);
            kptr += 32 * (long)E_;

            v8f s[2];
            s[0] = __builtin_amdgcn_wmma_f32_16x16x32_f16(false, qa[0], false, kv00, (short)0,
                       (v8f){0.f,0.f,0.f,0.f,0.f,0.f,0.f,0.f}, false, false);
            s[1] = __builtin_amdgcn_wmma_f32_16x16x32_f16(false, qa[0], false, kv10, (short)0,
                       (v8f){0.f,0.f,0.f,0.f,0.f,0.f,0.f,0.f}, false, false);
            s[0] = __builtin_amdgcn_wmma_f32_16x16x32_f16(false, qa[1], false, kv01, (short)0, s[0], false, false);
            s[1] = __builtin_amdgcn_wmma_f32_16x16x32_f16(false, qa[1], false, kv11, (short)0, s[1], false, false);
#pragma unroll
            for (int half = 0; half < 2; ++half)
#pragma unroll
                for (int r = 0; r < 8; ++r) s[half][r] *= scale;

            // causal mask only needed on the boundary tile (wave-uniform branch)
            if (ks + 32 > m0) {
#pragma unroll
                for (int r = 0; r < 8; ++r) {
                    int mg = m0 + r + hi * 8;
                    if (ks + lr > mg)      s[0][r] = -1e30f;
                    if (ks + 16 + lr > mg) s[1][r] = -1e30f;
                }
            }
            // online softmax per row: only the row-max needs a cross-lane reduction
#pragma unroll
            for (int r = 0; r < 8; ++r) {
                float rmax = fmaxf(s[0][r], s[1][r]);
#pragma unroll
                for (int off = 1; off < 16; off <<= 1) rmax = fmaxf(rmax, __shfl_xor(rmax, off, 32));
                float mnew = fmaxf(mrow[r], rmax);
                float alpha = __expf(mrow[r] - mnew);
                mrow[r] = mnew;
                float p0 = __expf(s[0][r] - mnew);
                float p1 = __expf(s[1][r] - mnew);
#pragma unroll
                for (int t = 0; t < 9; ++t) acc[t][r] *= alpha;
                lds_p[r + hi * 8][lr]      = (f16)p0;
                lds_p[r + hi * 8][16 + lr] = (f16)p1;
            }
            __syncthreads();
            // reshape P (C-layout in LDS) into an A-fragment
            v16h pa;
#pragma unroll
            for (int j = 0; j < 16; ++j) {
                int kk = j + (hi ? 16 : 0);
                pa[j] = lds_p[lr][kk];
            }
            __syncthreads();
            // PV: 8 n-tiles of 16 over head_dim 128; V^T gives contiguous B-fragments
#pragma unroll
            for (int t = 0; t < 8; ++t) {
                v16h bv = *(const v16h*)(vptr + (long)t * 16 * T_);
                acc[t] = __builtin_amdgcn_wmma_f32_16x16x32_f16(false, pa, false, bv, (short)0, acc[t], false, false);
            }
            vptr += 32;
            // row-sum tile: P @ ones -> every lane's element r holds sum_k P[row r][k]
            acc[8] = __builtin_amdgcn_wmma_f32_16x16x32_f16(false, pa, false, bones, (short)0, acc[8], false, false);
        }

        if (sel == 0) {
            // park normalized branch-1 output in LDS (each lane touches only its own slots)
#pragma unroll
            for (int r = 0; r < 8; ++r) {
                float inv = 1.0f / acc[8][r];
#pragma unroll
                for (int t = 0; t < 8; ++t)
                    lds_o1[r + hi * 8][t * 16 + lr] = acc[t][r] * inv;
            }
            __syncthreads();
        } else {
            // differential combine + RMS norm over head dim (128) + g scale
#pragma unroll
            for (int r = 0; r < 8; ++r) {
                float inv = 1.0f / acc[8][r];
                float av[8];
                float ssq = 0.f;
#pragma unroll
                for (int t = 0; t < 8; ++t) {
                    av[t] = lds_o1[r + hi * 8][t * 16 + lr] - lam * (acc[t][r] * inv);
                    ssq += av[t] * av[t];
                }
#pragma unroll
                for (int off = 1; off < 16; off <<= 1) ssq += __shfl_xor(ssq, off, 32);
                float rms = rsqrtf(ssq * (1.0f / 128.0f) + 1e-5f);
                int mg = m0 + r + hi * 8;
                f16* dst = a16 + (long)(b * T_ + mg) * E_ + h * HD_;
#pragma unroll
                for (int t = 0; t < 8; ++t) {
                    int n = t * 16 + lr;
                    dst[n] = (f16)(av[t] * rms * g[n] * ONE_MINUS_LAMBDA);
                }
            }
        }
    }
}

// ---------------------------------------------------------------------------------------------
extern "C" void kernel_launch(void* const* d_in, const int* in_sizes, int n_in,
                              void* d_out, int out_size, void* d_ws, size_t ws_size,
                              hipStream_t stream) {
    const float* x   = (const float*)d_in[0];
    const float* wq  = (const float*)d_in[1];
    const float* wk  = (const float*)d_in[2];
    const float* wv  = (const float*)d_in[3];
    const float* wo  = (const float*)d_in[4];
    const float* lq1 = (const float*)d_in[5];
    const float* lk1 = (const float*)d_in[6];
    const float* lq2 = (const float*)d_in[7];
    const float* lk2 = (const float*)d_in[8];
    const float* g   = (const float*)d_in[9];

    char* p = (char*)d_ws;
    auto carve = [&](size_t bytes) {
        void* r = (void*)p;
        p += (bytes + 255) & ~(size_t)255;
        return r;
    };
    const size_t tok_bytes = (size_t)NTOK * E_ * sizeof(f16);   // 8 MB
    const size_t w_bytes   = (size_t)E_ * E_ * sizeof(f16);     // 2 MB

    f16* x16  = (f16*)carve(tok_bytes);
    f16* wqT  = (f16*)carve(w_bytes);
    f16* wkT  = (f16*)carve(w_bytes);
    f16* wvT  = (f16*)carve(w_bytes);
    f16* woT  = (f16*)carve(w_bytes);
    f16* q16  = (f16*)carve(tok_bytes);
    f16* k16  = (f16*)carve(tok_bytes);
    f16* vt16 = (f16*)carve(tok_bytes);
    f16* a16  = (f16*)carve(tok_bytes);
    (void)ws_size;

    const int nTok = NTOK * E_;     // 4,194,304
    const int nW   = E_ * E_;       // 1,048,576

    // 1) convert x
    cvt_f32_f16<<<nTok / 256, 256, 0, stream>>>(x, x16, nTok);
    // 2) transpose+convert weights
    transpose_cvt_w<<<nW / 256, 256, 0, stream>>>(wq, wqT);
    transpose_cvt_w<<<nW / 256, 256, 0, stream>>>(wk, wkT);
    transpose_cvt_w<<<nW / 256, 256, 0, stream>>>(wv, wvT);
    transpose_cvt_w<<<nW / 256, 256, 0, stream>>>(wo, woT);

    // 3) QKV GEMMs: 32x64 tile per wave -> (4096/32)*(1024/64) = 2048 wave-tiles / 8 = 256 blocks
    const int gemm_blocks = (NTOK / 32) * (E_ / 64) / 8;
    gemm_wmma<0><<<gemm_blocks, 256, 0, stream>>>(x16, wqT, q16, NTOK, E_, E_);
    gemm_wmma<0><<<gemm_blocks, 256, 0, stream>>>(x16, wkT, k16, NTOK, E_, E_);
    // V GEMM writes directly into the transposed (B,H,128,T) layout
    gemm_wmma<2><<<gemm_blocks, 256, 0, stream>>>(x16, wvT, vt16, NTOK, E_, E_);

    // 4) fused differential flash attention (+diff, RMS norm, g, (1-lambda_init))
    dim3 agrid(T_ / 16, B_ * H_);
    diff_attn<<<agrid, 32, 0, stream>>>(q16, k16, vt16, a16, lq1, lk1, lq2, lk2, g);

    // 5) output projection -> f32 d_out
    gemm_wmma<1><<<gemm_blocks, 256, 0, stream>>>(a16, woT, (float*)d_out, NTOK, E_, E_);
}
